// HyperComputeCore_59253368815913
// MI455X (gfx1250) — matrix operand order repaired and checked by reading, not compiled, not run on hardware
//
#include <hip/hip_runtime.h>
#include <hip/hip_bf16.h>

// Problem constants (reference: B=2, C=128, H=W=64)
#define BSZ 2
#define CCH 128
#define NN  4096

typedef __attribute__((ext_vector_type(16))) __bf16 v16bf;
typedef __attribute__((ext_vector_type(8)))  float  v8f;

union FragU { uint4 q[2]; v16bf v; };

// A-matrix 16x32 bf16 fragment (row-major source). `row` points at the lane's
// row + current K offset. Lanes 0-15: K 0-15; lanes 16-31: K 16-31.
static __device__ inline v16bf frag_a16(const __hip_bfloat16* row, int lhalf) {
  FragU u;
  u.q[0] = *reinterpret_cast<const uint4*>(row + lhalf * 16);
  u.q[1] = *reinterpret_cast<const uint4*>(row + 8 + lhalf * 16);
  return u.v;
}

// B-matrix 32x16 bf16 fragment: lane holds 16 consecutive K for its column.
// `p` points at 32 contiguous bytes (K-major source row).
static __device__ inline v16bf frag_b16(const __hip_bfloat16* p) {
  FragU u;
  u.q[0] = *reinterpret_cast<const uint4*>(p);
  u.q[1] = *reinterpret_cast<const uint4*>(p + 8);
  return u.v;
}

static __device__ inline v8f wmma_bf16(v16bf a, v16bf b, v8f c) {
  // D = A*B + C, f32 accumulate; emits v_wmma_f32_16x16x32_bf16
  return __builtin_amdgcn_wmma_f32_16x16x32_bf16(false, a, false, b, (short)0, c,
                                                 false, false);
}

// LDS byte offset of a generic pointer that provably points into LDS.
static __device__ inline unsigned lds_offset(void* p) {
  return (unsigned)(unsigned long long)(__attribute__((address_space(3))) void*)p;
}

// Stage one 64B row (4 x b128) global -> LDS via the CDNA5 async DMA path.
// IOFFSET is added to BOTH the LDS and global addresses (ISA 15.18.3), so a
// single base pair covers the whole row. Tracked by ASYNCcnt.
#define ASYNC_ROW64B(ldsaddr, gaddr)                                           \
  do {                                                                         \
    asm volatile("global_load_async_to_lds_b128 %0, %1, off"                   \
                 ::"v"(ldsaddr), "v"(gaddr) : "memory");                       \
    asm volatile("global_load_async_to_lds_b128 %0, %1, off offset:16"         \
                 ::"v"(ldsaddr), "v"(gaddr) : "memory");                       \
    asm volatile("global_load_async_to_lds_b128 %0, %1, off offset:32"         \
                 ::"v"(ldsaddr), "v"(gaddr) : "memory");                       \
    asm volatile("global_load_async_to_lds_b128 %0, %1, off offset:48"         \
                 ::"v"(ldsaddr), "v"(gaddr) : "memory");                       \
  } while (0)

#define WAIT_ASYNC0() asm volatile("s_wait_asynccnt 0" ::: "memory")

// ---------------------------------------------------------------------------
// K0: x [B][C][N] f32 -> xf hi/lo bf16 split, node-major [B][N][C]
// ---------------------------------------------------------------------------
__global__ void k0_split_transpose(const float* __restrict__ x,
                                   __hip_bfloat16* __restrict__ xh,
                                   __hip_bfloat16* __restrict__ xl) {
  __shared__ float t[32][33];
  int b = blockIdx.z;
  int n0 = blockIdx.x * 32, c0 = blockIdx.y * 32;
  const float* xb = x + (size_t)b * CCH * NN;
  for (int i = threadIdx.y; i < 32; i += 8)
    t[i][threadIdx.x] = xb[(size_t)(c0 + i) * NN + n0 + threadIdx.x];
  __syncthreads();
  __hip_bfloat16* xhb = xh + (size_t)b * NN * CCH;
  __hip_bfloat16* xlb = xl + (size_t)b * NN * CCH;
  for (int i = threadIdx.y; i < 32; i += 8) {
    float v = t[threadIdx.x][i];
    __hip_bfloat16 hi = __float2bfloat16(v);
    float lo = v - __bfloat162float(hi);
    size_t o = (size_t)(n0 + i) * CCH + c0 + threadIdx.x;
    xhb[o] = hi;
    xlb[o] = __float2bfloat16(lo);
  }
}

// K0b: squared row norms sq[b][n] = sum_c x^2
__global__ void k0b_sq(const float* __restrict__ x, float* __restrict__ sq) {
  int n = blockIdx.x * 256 + threadIdx.x;
  int b = blockIdx.y;
  const float* xb = x + (size_t)b * CCH * NN + n;
  float s = 0.f;
  for (int c = 0; c < CCH; ++c) { float v = xb[(size_t)c * NN]; s += v * v; }
  sq[b * NN + n] = s;
}

// K0c: W_fc [C][C] f32 -> hi/lo bf16 split
__global__ void k0c_wsplit(const float* __restrict__ W,
                           __hip_bfloat16* __restrict__ wh,
                           __hip_bfloat16* __restrict__ wl) {
  int i = blockIdx.x * 256 + threadIdx.x;
  float v = W[i];
  __hip_bfloat16 h = __float2bfloat16(v);
  wh[i] = h;
  wl[i] = __float2bfloat16(v - __bfloat162float(h));
}

// ---------------------------------------------------------------------------
// K1: Gram (bf16x3 split) -> distance threshold -> Hg bf16 [B][N][N]
// One 16x16 tile per wave; K=C=128 in 4 steps of 32.
// ---------------------------------------------------------------------------
__global__ void k1_gram(const __hip_bfloat16* __restrict__ xh,
                        const __hip_bfloat16* __restrict__ xl,
                        const float* __restrict__ sq,
                        __hip_bfloat16* __restrict__ Hg) {
  int b = blockIdx.z;
  int wave = threadIdx.x >> 5, lane = threadIdx.x & 31;
  int lhalf = lane >> 4, lmod = lane & 15;
  int i0 = blockIdx.x * 16;
  int j0 = (blockIdx.y * 8 + wave) * 16;
  const __hip_bfloat16* xhb = xh + (size_t)b * NN * CCH;
  const __hip_bfloat16* xlb = xl + (size_t)b * NN * CCH;

  v8f acc = {0.f, 0.f, 0.f, 0.f, 0.f, 0.f, 0.f, 0.f};
  const __hip_bfloat16* arowh = xhb + (size_t)(i0 + lmod) * CCH;
  const __hip_bfloat16* arowl = xlb + (size_t)(i0 + lmod) * CCH;
  const __hip_bfloat16* browh = xhb + (size_t)(j0 + lmod) * CCH + lhalf * 16;
  const __hip_bfloat16* browl = xlb + (size_t)(j0 + lmod) * CCH + lhalf * 16;
#pragma unroll
  for (int k = 0; k < CCH; k += 32) {
    v16bf ah = frag_a16(arowh + k, lhalf);
    v16bf al = frag_a16(arowl + k, lhalf);
    v16bf bh = frag_b16(browh + k);
    v16bf bl = frag_b16(browl + k);
    acc = wmma_bf16(ah, bh, acc);
    acc = wmma_bf16(ah, bl, acc);
    acc = wmma_bf16(al, bh, acc);
  }
  const float* sqb = sq + b * NN;
  float sqj = sqb[j0 + lmod];
  __hip_bfloat16* Hb = Hg + (size_t)b * NN * NN;
#pragma unroll
  for (int r = 0; r < 8; ++r) {
    int i = i0 + r + lhalf * 8;
    float d2 = sqb[i] + sqj - 2.f * acc[r];
    // dist < 8  <=>  d2 < 64 (clip-to-0 cases also satisfy)
    Hb[(size_t)i * NN + j0 + lmod] = __float2bfloat16(d2 < 64.f ? 1.f : 0.f);
  }
}

// K1b: degrees -> invdeg[b][i] = 1/rowsum(Hg) (symmetric: row==col sums)
__global__ void k1b_deg(const __hip_bfloat16* __restrict__ Hg,
                        float* __restrict__ invdeg) {
  int row = blockIdx.x, b = blockIdx.y;
  const __hip_bfloat16* p = Hg + ((size_t)b * NN + row) * NN;
  float s = 0.f;
  for (int j = threadIdx.x; j < NN; j += 256) s += __bfloat162float(p[j]);
  __shared__ float red[256];
  red[threadIdx.x] = s;
  __syncthreads();
  for (int o = 128; o > 0; o >>= 1) {
    if (threadIdx.x < o) red[threadIdx.x] += red[threadIdx.x + o];
    __syncthreads();
  }
  if (threadIdx.x == 0) invdeg[b * NN + row] = red[0] > 0.f ? 1.f / red[0] : 0.f;
}

// ---------------------------------------------------------------------------
// K2: y = xf @ W^T + b (bf16x3 split). Wave: 16 rows x 128 cols (8 accums).
// Writes y f32 [B][N][C] (residual) and yT bf16 [B][C][N] (aggregation B-op).
// ---------------------------------------------------------------------------
__global__ void k2_ygemm(const __hip_bfloat16* __restrict__ xh,
                         const __hip_bfloat16* __restrict__ xl,
                         const __hip_bfloat16* __restrict__ wh,
                         const __hip_bfloat16* __restrict__ wl,
                         const float* __restrict__ bias,
                         float* __restrict__ y,
                         __hip_bfloat16* __restrict__ yT) {
  int b = blockIdx.y;
  int wave = threadIdx.x >> 5, lane = threadIdx.x & 31;
  int lhalf = lane >> 4, lmod = lane & 15;
  int i0 = (blockIdx.x * 8 + wave) * 16;
  const __hip_bfloat16* xhb = xh + (size_t)b * NN * CCH;
  const __hip_bfloat16* xlb = xl + (size_t)b * NN * CCH;

  v8f acc[8];
#pragma unroll
  for (int f = 0; f < 8; ++f) acc[f] = (v8f){0.f, 0.f, 0.f, 0.f, 0.f, 0.f, 0.f, 0.f};

  const __hip_bfloat16* arowh = xhb + (size_t)(i0 + lmod) * CCH;
  const __hip_bfloat16* arowl = xlb + (size_t)(i0 + lmod) * CCH;
#pragma unroll
  for (int k = 0; k < CCH; k += 32) {
    v16bf ah = frag_a16(arowh + k, lhalf);
    v16bf al = frag_a16(arowl + k, lhalf);
#pragma unroll
    for (int f = 0; f < 8; ++f) {
      size_t wo = (size_t)(f * 16 + lmod) * CCH + k + lhalf * 16;
      v16bf bh = frag_b16(wh + wo);
      v16bf bl = frag_b16(wl + wo);
      acc[f] = wmma_bf16(ah, bh, acc[f]);
      acc[f] = wmma_bf16(ah, bl, acc[f]);
      acc[f] = wmma_bf16(al, bh, acc[f]);
    }
  }
  float* yb = y + (size_t)b * NN * CCH;
  __hip_bfloat16* yTb = yT + (size_t)b * CCH * NN;
#pragma unroll
  for (int f = 0; f < 8; ++f) {
    int d = f * 16 + lmod;
    float bv = bias[d];
#pragma unroll
    for (int r = 0; r < 8; ++r) {
      int i = i0 + r + lhalf * 8;
      float v = acc[f][r] + bv;
      yb[(size_t)i * CCH + d] = v;
      yTb[(size_t)d * NN + i] = __float2bfloat16(v);
    }
  }
}

// ---------------------------------------------------------------------------
// K3: aggregation out[i][:] = invdeg[i] * sum_j Hg[i][j] * Bm[:,j]
// Wave: 16 rows x 128 channels. Bm K-slab (128 x 32) staged via
// GLOBAL_LOAD_ASYNC_TO_LDS_B128 into double-buffered LDS (padded row stride
// 40 elems = 80B for conflict-free b128 reads), shared by 4 waves. DMA for
// slab k+1 overlaps the 8 WMMAs on slab k; one barrier per K-step:
//   reads of buf[p] precede barrier i; writes of buf[p^1] precede barrier i;
//   buf[p] is only rewritten after barrier i.
// mode 0: write ET bf16 [C][N].   mode 1: write xeT f32 [C][N] (+ y residual).
// ---------------------------------------------------------------------------
__global__ void k3_agg(const __hip_bfloat16* __restrict__ Hg,
                       const __hip_bfloat16* __restrict__ Bm,
                       const float* __restrict__ invdeg,
                       const float* __restrict__ yres,
                       __hip_bfloat16* __restrict__ outT_bf,
                       float* __restrict__ outT_f,
                       int mode) {
  __shared__ __hip_bfloat16 ldsB[2 * CCH * 40];  // 2 x 10240 B
  int b = blockIdx.y;
  int wave = threadIdx.x >> 5, lane = threadIdx.x & 31;
  int lhalf = lane >> 4, lmod = lane & 15;
  int i0 = (blockIdx.x * 4 + wave) * 16;
  const __hip_bfloat16* Hb = Hg + (size_t)b * NN * NN;
  const __hip_bfloat16* Bb = Bm + (size_t)b * CCH * NN;

  v8f acc[8];
#pragma unroll
  for (int f = 0; f < 8; ++f) acc[f] = (v8f){0.f, 0.f, 0.f, 0.f, 0.f, 0.f, 0.f, 0.f};

  const __hip_bfloat16* arow = Hb + (size_t)(i0 + lmod) * NN;
  int t = threadIdx.x;  // 0..127 == channel row for cooperative staging
  unsigned myLds = lds_offset(ldsB) + (unsigned)t * 80u;  // row base, buffer 0
  unsigned long long gRow = (unsigned long long)(Bb + (size_t)t * NN);

  // Prologue: DMA slab j0=0 into buffer 0.
  ASYNC_ROW64B(myLds, gRow);
  WAIT_ASYNC0();
  __syncthreads();

  int p = 0;
  for (int j0 = 0; j0 < NN; j0 += 32) {
    if (j0 + 32 < NN) {  // kick DMA for next slab into the other buffer
      unsigned nl = myLds + (p ? 0u : 10240u);
      ASYNC_ROW64B(nl, gRow + (unsigned long long)(j0 + 32) * 2u);
    }
    const __hip_bfloat16* cur = ldsB + p * (CCH * 40);
    v16bf a = frag_a16(arow + j0, lhalf);
    if (j0 + 32 < NN) __builtin_prefetch(arow + j0 + 32, 0, 3);
#pragma unroll
    for (int f = 0; f < 8; ++f) {
      const __hip_bfloat16* lp = cur + (f * 16 + lmod) * 40 + lhalf * 16;
      acc[f] = wmma_bf16(a, frag_b16(lp), acc[f]);
    }
    WAIT_ASYNC0();     // my DMA into next buffer has landed
    __syncthreads();   // everyone done reading cur & done staging next
    p ^= 1;
  }

  const float* idb = invdeg + b * NN;
#pragma unroll
  for (int f = 0; f < 8; ++f) {
    int c = f * 16 + lmod;
#pragma unroll
    for (int r = 0; r < 8; ++r) {
      int i = i0 + r + lhalf * 8;
      float v = acc[f][r] * idb[i];
      if (mode == 0) {
        outT_bf[(size_t)b * CCH * NN + (size_t)c * NN + i] = __float2bfloat16(v);
      } else {
        v += yres[(size_t)b * NN * CCH + (size_t)i * CCH + c];
        outT_f[(size_t)b * CCH * NN + (size_t)c * NN + i] = v;
      }
    }
  }
}

// ---------------------------------------------------------------------------
// K5a: per-channel batch stats -> fused scale/shift.  K5b: normalize + SiLU.
// ---------------------------------------------------------------------------
__global__ void k5a_stats(const float* __restrict__ xeT,
                          const float* __restrict__ gamma,
                          const float* __restrict__ beta,
                          float* __restrict__ scale,
                          float* __restrict__ shift) {
  int c = blockIdx.x;
  float s = 0.f, s2 = 0.f;
  for (int b = 0; b < BSZ; ++b) {
    const float* p = xeT + (size_t)b * CCH * NN + (size_t)c * NN;
    for (int n = threadIdx.x; n < NN; n += 256) {
      float v = p[n];
      s += v;
      s2 += v * v;
    }
  }
  __shared__ float rs[256], rs2[256];
  rs[threadIdx.x] = s;
  rs2[threadIdx.x] = s2;
  __syncthreads();
  for (int o = 128; o > 0; o >>= 1) {
    if (threadIdx.x < o) {
      rs[threadIdx.x] += rs[threadIdx.x + o];
      rs2[threadIdx.x] += rs2[threadIdx.x + o];
    }
    __syncthreads();
  }
  if (threadIdx.x == 0) {
    float inv = 1.f / (float)(BSZ * NN);
    float mean = rs[0] * inv;
    float var = rs2[0] * inv - mean * mean;
    float sc = gamma[c] * rsqrtf(var + 1e-5f);
    scale[c] = sc;
    shift[c] = beta[c] - mean * sc;
  }
}

__global__ void k5b_out(const float* __restrict__ xeT,
                        const float* __restrict__ scale,
                        const float* __restrict__ shift,
                        float* __restrict__ out) {
  size_t idx = (size_t)blockIdx.x * 256 + threadIdx.x;
  int c = (int)((idx / NN) % CCH);
  float v = xeT[idx];
  float xn = v * scale[c] + shift[c];
  out[idx] = xn / (1.f + __expf(-xn));  // xn * sigmoid(xn)
}

// ---------------------------------------------------------------------------
extern "C" void kernel_launch(void* const* d_in, const int* in_sizes, int n_in,
                              void* d_out, int out_size, void* d_ws, size_t ws_size,
                              hipStream_t stream) {
  const float* x     = (const float*)d_in[0];  // [B][C][H][W]
  const float* W     = (const float*)d_in[1];  // [C][C] (row d, col c)
  const float* bias  = (const float*)d_in[2];  // [C]
  const float* gamma = (const float*)d_in[3];  // [C]
  const float* beta  = (const float*)d_in[4];  // [C]
  float* out = (float*)d_out;

  char* w = (char*)d_ws;
  auto alloc = [&](size_t bytes) -> char* {
    char* p = w;
    w += (bytes + 255) & ~(size_t)255;
    return p;
  };
  __hip_bfloat16* xh  = (__hip_bfloat16*)alloc((size_t)BSZ * NN * CCH * 2);
  __hip_bfloat16* xl  = (__hip_bfloat16*)alloc((size_t)BSZ * NN * CCH * 2);
  float*          sqv = (float*)alloc((size_t)BSZ * NN * 4);
  __hip_bfloat16* wh  = (__hip_bfloat16*)alloc((size_t)CCH * CCH * 2);
  __hip_bfloat16* wl  = (__hip_bfloat16*)alloc((size_t)CCH * CCH * 2);
  float*          y   = (float*)alloc((size_t)BSZ * NN * CCH * 4);
  __hip_bfloat16* yT  = (__hip_bfloat16*)alloc((size_t)BSZ * CCH * NN * 2);
  __hip_bfloat16* Hg  = (__hip_bfloat16*)alloc((size_t)BSZ * NN * NN * 2);  // 64 MB
  float*          idg = (float*)alloc((size_t)BSZ * NN * 4);
  __hip_bfloat16* ET  = (__hip_bfloat16*)alloc((size_t)BSZ * CCH * NN * 2);
  float*          xeT = (float*)alloc((size_t)BSZ * CCH * NN * 4);
  float*          scl = (float*)alloc((size_t)CCH * 4);
  float*          sft = (float*)alloc((size_t)CCH * 4);

  k0_split_transpose<<<dim3(NN / 32, CCH / 32, BSZ), dim3(32, 8, 1), 0, stream>>>(x, xh, xl);
  k0b_sq<<<dim3(NN / 256, BSZ), 256, 0, stream>>>(x, sqv);
  k0c_wsplit<<<(CCH * CCH) / 256, 256, 0, stream>>>(W, wh, wl);
  k2_ygemm<<<dim3(NN / 128, BSZ), 256, 0, stream>>>(xh, xl, wh, wl, bias, y, yT);
  k1_gram<<<dim3(NN / 16, NN / 128, BSZ), 256, 0, stream>>>(xh, xl, sqv, Hg);
  k1b_deg<<<dim3(NN, BSZ), 256, 0, stream>>>(Hg, idg);
  k3_agg<<<dim3(NN / 64, BSZ), 128, 0, stream>>>(Hg, yT, idg, nullptr, ET, nullptr, 0);
  k3_agg<<<dim3(NN / 64, BSZ), 128, 0, stream>>>(Hg, ET, idg, y, nullptr, xeT, 1);
  k5a_stats<<<CCH, 256, 0, stream>>>(xeT, gamma, beta, scl, sft);
  k5b_out<<<(BSZ * CCH * NN) / 256, 256, 0, stream>>>(xeT, scl, sft, out);
}
